// TimeAttention_61426622267526
// MI455X (gfx1250) — matrix operand order repaired and checked by reading, not compiled, not run on hardware
//
#include <hip/hip_runtime.h>

// ---------------------------------------------------------------------------
// TimeAttention for MI455X (gfx1250, wave32, WMMA bf16 16x16x32, f32 accum)
// B=128, N=15 (slots=16), T=128, H=128
// ---------------------------------------------------------------------------

#define T_DIM 128
#define H_DIM 128
#define SLOTS 16
#define LD    136                      // LDS row stride in bf16 elems (272 B)
#define TILE_BYTES (128 * LD * 2)      // one 128x128 bf16 LDS tile = 34816 B

typedef __attribute__((ext_vector_type(16))) __bf16 v16bf;
typedef __attribute__((ext_vector_type(8)))  float  v8f;

union V16U { v16bf v; uint4 q[2]; };

static __device__ __forceinline__ __bf16 f2bf(float f) { return (__bf16)f; }

#if defined(__has_builtin)
#if __has_builtin(__builtin_amdgcn_global_load_async_to_lds_b128)
#define USE_ASYNC_LDS 1
#endif
#endif
#ifndef USE_ASYNC_LDS
#define USE_ASYNC_LDS 0
#endif

#if USE_ASYNC_LDS
// Builtin signature (from compiler diagnostic): (v4i addrspace(1)*,
// v4i addrspace(3)*, imm int offset, imm int cpol), v4i = vector_size(16) int.
typedef int v4i_t __attribute__((__vector_size__(16)));
typedef __attribute__((address_space(1))) v4i_t* gptr_v4i;
typedef __attribute__((address_space(3))) v4i_t* lptr_v4i;

static __device__ __forceinline__ void wait_async0() {
#if __has_builtin(__builtin_amdgcn_s_wait_asynccnt)
  __builtin_amdgcn_s_wait_asynccnt(0);
#else
  asm volatile("s_wait_asynccnt 0" ::: "memory");
#endif
}
#endif

// Copy a row-major [128][128] bf16 tile from global to LDS (row stride LD).
// Prefer the CDNA5 async global->LDS DMA path (ASYNCcnt, no VGPR round-trip);
// fall back to a dword copy when the builtin is unavailable.
static __device__ __forceinline__ void stage_tile_bf16(const __bf16* __restrict__ g,
                                                       __bf16* l, int tid) {
#if USE_ASYNC_LDS
  for (int i = tid; i < 2048; i += 256) {          // 2048 chunks of 16 B
    int r = i >> 4, c = i & 15;
    char* gp = (char*)g + i * 16;                  // drop const for builtin
    char* lp = (char*)l + r * (LD * 2) + c * 16;
    __builtin_amdgcn_global_load_async_to_lds_b128((gptr_v4i)gp, (lptr_v4i)lp,
                                                   0, 0);
  }
#else
  const unsigned* s = (const unsigned*)g;
  unsigned* d = (unsigned*)l;
  for (int i = tid; i < T_DIM * H_DIM / 2; i += 256) {
    int r = i >> 6, dw = i & 63;
    d[r * (LD / 2) + dw] = s[i];
  }
#endif
}

// A-operand (16x32 bf16, M x K) gather from row-major LDS tile.
// ISA layout: lanes 0-15 -> K 0..7 & 16..23; lanes 16-31 -> K 8..15 & 24..31.
static __device__ __forceinline__ v16bf load_a_tile(const __bf16* s, int row0,
                                                    int k0, int lane) {
  int r  = row0 + (lane & 15);
  int kb = (lane >> 4) << 3;           // 0 or 8
  const __bf16* p = s + r * LD + k0;
  V16U u;
  u.q[0] = *(const uint4*)(p + kb);         // K = k0+kb .. +7
  u.q[1] = *(const uint4*)(p + 16 + kb);    // K = k0+16+kb .. +7
  return u.v;
}

// B-operand (32x16 bf16, K x N) gather. Matrix stored row-major as [N][K]
// (one row per output column). Lane L holds column N = L%16 and
// K = (L/16)*8 .. +7 and 16+(L/16)*8 .. +7 (same order as A).
static __device__ __forceinline__ v16bf load_b_tile(const __bf16* s, int n0,
                                                    int k0, int lane) {
  int r  = n0 + (lane & 15);
  int kb = (lane >> 4) << 3;           // 0 or 8
  const __bf16* p = s + r * LD + k0 + kb;
  V16U u;
  u.q[0] = *(const uint4*)(p);
  u.q[1] = *(const uint4*)(p + 16);
  return u.v;
}

static __device__ __forceinline__ v8f wmma_bf16(v16bf a, v16bf b, v8f c) {
  return __builtin_amdgcn_wmma_f32_16x16x32_bf16(false, a, false, b,
                                                 (short)0, c, false, false);
}

// One wave computes a 16x128 strip: D[m0:m0+16, :] = A[m0:m0+16, 0:128] * B.
// sB is row-major [N][K] (B-operand convention above). K = 128.
static __device__ __forceinline__ void gemm_strip(const __bf16* sA, int m0,
                                                  const __bf16* sB,
                                                  v8f acc[8], int lane) {
#pragma unroll
  for (int kk = 0; kk < 4; ++kk) {
    v16bf a = load_a_tile(sA, m0, kk * 32, lane);
#pragma unroll
    for (int nt = 0; nt < 8; ++nt) {
      v16bf b = load_b_tile(sB, nt * 16, kk * 32, lane);
      acc[nt] = wmma_bf16(a, b, acc[nt]);
    }
  }
}

// ---------------------------------------------------------------------------
// prep: Wqbt[p][h] = sum_o Wb[o][p] * Wq[o][h]  (so qb = node @ Wqb), plus
// bf16 conversion of Wk, Wv. Tiny (2 MFLOP), one block.
// ---------------------------------------------------------------------------
__global__ void prep_kernel(const float* __restrict__ Wq,
                            const float* __restrict__ Wk,
                            const float* __restrict__ Wv,
                            const float* __restrict__ Wb,
                            __bf16* __restrict__ wk16,
                            __bf16* __restrict__ wv16,
                            __bf16* __restrict__ wqbt) {
  int tid = threadIdx.x;
  for (int i = tid; i < H_DIM * H_DIM; i += 256) {
    wk16[i] = f2bf(Wk[i]);
    wv16[i] = f2bf(Wv[i]);
  }
  for (int i = tid; i < H_DIM * H_DIM; i += 256) {
    int p = i >> 7, h = i & 127;
    float acc = 0.f;
    for (int o = 0; o < H_DIM; ++o)
      acc += Wb[o * H_DIM + p] * Wq[o * H_DIM + h];
    wqbt[i] = f2bf(acc);               // row-major [p][h]
  }
}

// ---------------------------------------------------------------------------
// qb: qb16[b] = node[b] (128x128) @ Wqb, bf16 out. One block per b.
// ---------------------------------------------------------------------------
__global__ void qb_kernel(const float* __restrict__ node,
                          const __bf16* __restrict__ wqbt,
                          __bf16* __restrict__ qb16) {
  extern __shared__ char smem[];
  __bf16* sN = (__bf16*)(smem);
  __bf16* sW = (__bf16*)(smem + TILE_BYTES);

  int b = blockIdx.x, tid = threadIdx.x;
  int wave = tid >> 5, lane = tid & 31;

  stage_tile_bf16(wqbt, sW, tid);      // async DMA (or fallback copy)

  const float* src = node + (size_t)b * T_DIM * H_DIM;
  for (int i = tid; i < T_DIM * H_DIM / 2; i += 256) {
    int e = i * 2, r = e >> 7, c = e & 127;
    float2 f = *(const float2*)(src + e);
    sN[r * LD + c]     = f2bf(f.x);
    sN[r * LD + c + 1] = f2bf(f.y);
  }
#if USE_ASYNC_LDS
  wait_async0();
#endif
  __syncthreads();

  v8f acc[8];
#pragma unroll
  for (int nt = 0; nt < 8; ++nt) acc[nt] = (v8f){0,0,0,0,0,0,0,0};
  int m0 = wave * 16;
  gemm_strip(sN, m0, sW, acc, lane);

  int col0 = lane & 15, rh = (lane >> 4) * 8;
#pragma unroll
  for (int nt = 0; nt < 8; ++nt)
#pragma unroll
    for (int j = 0; j < 8; ++j) {
      int t = m0 + j + rh, p = nt * 16 + col0;
      qb16[((size_t)b * T_DIM + t) * H_DIM + p] = f2bf(acc[nt][j]);
    }
}

// ---------------------------------------------------------------------------
// attn: one block per (b, slot). 8 waves x 16-row strips.
// ---------------------------------------------------------------------------
__global__ void attn_kernel(const float* __restrict__ node,
                            const float* __restrict__ neigh,
                            const unsigned char* __restrict__ mask,
                            const __bf16* __restrict__ wk16,
                            const __bf16* __restrict__ wv16,
                            const __bf16* __restrict__ qb16,
                            float* __restrict__ out_o,
                            float* __restrict__ out_a) {
  extern __shared__ char smem[];
  __bf16* sA  = (__bf16*)(smem);                  // KV bf16; later aliased = sS
  __bf16* sWk = (__bf16*)(smem + 1 * TILE_BYTES);
  __bf16* sWv = (__bf16*)(smem + 2 * TILE_BYTES);
  __bf16* sQB = (__bf16*)(smem + 3 * TILE_BYTES);
  __bf16* sK  = (__bf16*)(smem + 4 * TILE_BYTES); // K rows [s][p]
  __bf16* sVt = (__bf16*)(smem + 5 * TILE_BYTES); // V transposed [h][s]
  __bf16* sS  = sA;                               // A scores bf16 [t][s]

  int b = blockIdx.x >> 4, n = blockIdx.x & 15;
  int tid = threadIdx.x, wave = tid >> 5, lane = tid & 31;

  // ---- stage: async DMA the bf16 operands, convert KV f32->bf16 in VALU ---
  stage_tile_bf16(wk16, sWk, tid);
  stage_tile_bf16(wv16, sWv, tid);
  stage_tile_bf16(qb16 + (size_t)b * T_DIM * H_DIM, sQB, tid);

  const float* kv = (n == 0)
      ? node + (size_t)b * T_DIM * H_DIM
      : neigh + ((size_t)b * 15 + (n - 1)) * T_DIM * H_DIM;
  for (int i = tid; i < T_DIM * H_DIM / 2; i += 256) {
    int e = i * 2, r = e >> 7, c = e & 127;
    float2 f = *(const float2*)(kv + e);
    sA[r * LD + c]     = f2bf(f.x);
    sA[r * LD + c + 1] = f2bf(f.y);
  }
#if USE_ASYNC_LDS
  wait_async0();
#endif
  __syncthreads();

  int m0 = wave * 16, col0 = lane & 15, rh = (lane >> 4) * 8;
  v8f acc[8];

  // ---- K projection: K = KV @ Wk^T  -> sK rows ---------------------------
#pragma unroll
  for (int nt = 0; nt < 8; ++nt) acc[nt] = (v8f){0,0,0,0,0,0,0,0};
  gemm_strip(sA, m0, sWk, acc, lane);
#pragma unroll
  for (int nt = 0; nt < 8; ++nt)
#pragma unroll
    for (int j = 0; j < 8; ++j) {
      int t = m0 + j + rh, p = nt * 16 + col0;
      sK[t * LD + p] = f2bf(acc[nt][j]);
    }

  // ---- V projection: Vt = (KV @ Wv^T)^T -> sVt[h][s] ---------------------
#pragma unroll
  for (int nt = 0; nt < 8; ++nt) acc[nt] = (v8f){0,0,0,0,0,0,0,0};
  gemm_strip(sA, m0, sWv, acc, lane);
#pragma unroll
  for (int nt = 0; nt < 8; ++nt)
#pragma unroll
    for (int j = 0; j < 8; ++j) {
      int s = m0 + j + rh, h = nt * 16 + col0;
      sVt[h * LD + s] = f2bf(acc[nt][j]);
    }
  __syncthreads();

  // ---- scores: S = qb @ K^T  (B-operand = sK rows [s][p]) ----------------
#pragma unroll
  for (int nt = 0; nt < 8; ++nt) acc[nt] = (v8f){0,0,0,0,0,0,0,0};
  gemm_strip(sQB, m0, sK, acc, lane);

  // ---- mask + softmax (per row t), emit A (f32 global + bf16 LDS) --------
  const unsigned char* mrow = mask + ((size_t)b * T_DIM) * T_DIM;
  float* arow = out_a + (((size_t)b * SLOTS + n) * T_DIM) * T_DIM;
#pragma unroll
  for (int j = 0; j < 8; ++j) {
    int t = m0 + j + rh;
#pragma unroll
    for (int nt = 0; nt < 8; ++nt) {
      int s = nt * 16 + col0;
      if (mrow[t * T_DIM + s]) acc[nt][j] = -__builtin_inff();
    }
    float mx = -__builtin_inff();
#pragma unroll
    for (int nt = 0; nt < 8; ++nt) mx = fmaxf(mx, acc[nt][j]);
    for (int off = 1; off < 16; off <<= 1)
      mx = fmaxf(mx, __shfl_xor(mx, off, 32));
    float sum = 0.f;
#pragma unroll
    for (int nt = 0; nt < 8; ++nt) {
      float e = __expf(acc[nt][j] - mx);
      acc[nt][j] = e;
      sum += e;
    }
    for (int off = 1; off < 16; off <<= 1)
      sum += __shfl_xor(sum, off, 32);
    float inv = 1.f / sum;
#pragma unroll
    for (int nt = 0; nt < 8; ++nt) {
      int s = nt * 16 + col0;
      float a = acc[nt][j] * inv;
      arow[t * T_DIM + s] = a;          // A output (f32)
      sS[t * LD + s] = f2bf(a);         // A operand for A@V (own rows only)
    }
  }

  // ---- output: out = A @ V  (B-operand = sVt rows [h][s]) ----------------
#pragma unroll
  for (int nt = 0; nt < 8; ++nt) acc[nt] = (v8f){0,0,0,0,0,0,0,0};
  gemm_strip(sS, m0, sVt, acc, lane);

  float* orow = out_o + (((size_t)b * SLOTS + n) * T_DIM) * H_DIM;
#pragma unroll
  for (int nt = 0; nt < 8; ++nt)
#pragma unroll
    for (int j = 0; j < 8; ++j) {
      int t = m0 + j + rh, h = nt * 16 + col0;
      orow[t * H_DIM + h] = acc[nt][j];
    }
}

// ---------------------------------------------------------------------------
extern "C" void kernel_launch(void* const* d_in, const int* in_sizes, int n_in,
                              void* d_out, int out_size, void* d_ws,
                              size_t ws_size, hipStream_t stream) {
  const float* node  = (const float*)d_in[0];
  const float* neigh = (const float*)d_in[1];
  // d_in[2] = neighbors_number: unused by the reference computation
  const unsigned char* mask = (const unsigned char*)d_in[3];
  const float* Wq = (const float*)d_in[4];
  const float* Wk = (const float*)d_in[5];
  const float* Wv = (const float*)d_in[6];
  const float* Wb = (const float*)d_in[7];

  char* ws = (char*)d_ws;
  __bf16* wk16 = (__bf16*)(ws);
  __bf16* wv16 = (__bf16*)(ws + 32768);
  __bf16* wqbt = (__bf16*)(ws + 65536);
  __bf16* qb16 = (__bf16*)(ws + 98304);   // 128*128*128 bf16 = 4 MiB

  float* out_o = (float*)d_out;
  float* out_a = out_o + (size_t)128 * SLOTS * T_DIM * H_DIM;

  (void)hipFuncSetAttribute((const void*)qb_kernel,
                            hipFuncAttributeMaxDynamicSharedMemorySize,
                            2 * TILE_BYTES);
  (void)hipFuncSetAttribute((const void*)attn_kernel,
                            hipFuncAttributeMaxDynamicSharedMemorySize,
                            6 * TILE_BYTES);

  prep_kernel<<<1, 256, 0, stream>>>(Wq, Wk, Wv, Wb, wk16, wv16, wqbt);
  qb_kernel<<<128, 256, 2 * TILE_BYTES, stream>>>(node, wqbt, qb16);
  attn_kernel<<<128 * SLOTS, 256, 6 * TILE_BYTES, stream>>>(
      node, neigh, mask, wk16, wv16, qb16, out_o, out_a);
}
